// GNNModel_54855322305069
// MI455X (gfx1250) — matrix operand order repaired and checked by reading, not compiled, not run on hardware
//
#include <hip/hip_runtime.h>

#define HDIM 128
#define GDIM 64
#define CDIM 10

typedef __attribute__((ext_vector_type(16))) __bf16          v16bf;
typedef __attribute__((ext_vector_type(16))) unsigned short  v16us;
typedef __attribute__((ext_vector_type(8)))  unsigned short  v8us;
typedef __attribute__((ext_vector_type(8)))  float           v8f;

// ---------- bf16 split helpers (bit-exact, no __bf16 arithmetic) ----------
__device__ __forceinline__ unsigned short f2bf(float f) {
  unsigned int u = __float_as_uint(f);
  u += 0x7FFFu + ((u >> 16) & 1u);           // round-to-nearest-even
  return (unsigned short)(u >> 16);
}
__device__ __forceinline__ float bf2f(unsigned short h) {
  return __uint_as_float(((unsigned int)h) << 16);
}

// ---------- small elementwise kernels ----------
__global__ void k_zero(float* __restrict__ p, long n) {
  long t = (long)blockIdx.x * blockDim.x + threadIdx.x;
  if (t < n) p[t] = 0.0f;
}
__global__ void k_fill1(float* __restrict__ p, int n) {
  int t = blockIdx.x * blockDim.x + threadIdx.x;
  if (t < n) p[t] = 1.0f;                     // self-loop contributes 1 to degree
}
__global__ void k_deg(const int* __restrict__ col, float* __restrict__ deg, int E) {
  int t = blockIdx.x * blockDim.x + threadIdx.x;
  if (t < E) atomicAdd(&deg[col[t]], 1.0f);
}
__global__ void k_rsqrt(float* __restrict__ d, int n) {
  int t = blockIdx.x * blockDim.x + threadIdx.x;
  if (t < n) d[t] = rsqrtf(d[t]);             // deg >= 1 always (self-loops)
}

// transpose W[k][n] -> Wt[n][k], split into bf16 hi/lo
__global__ void k_splitW(const float* __restrict__ W,
                         unsigned short* __restrict__ thi,
                         unsigned short* __restrict__ tlo) {
  int t = blockIdx.x * blockDim.x + threadIdx.x;   // t over 128*128
  if (t >= HDIM * HDIM) return;
  int n = t >> 7, k = t & 127;
  float w = W[k * HDIM + n];
  unsigned short hi = f2bf(w);
  thi[t] = hi;
  tlo[t] = f2bf(w - bf2f(hi));
}
__global__ void k_splitX(const float* __restrict__ X,
                         unsigned short* __restrict__ xhi,
                         unsigned short* __restrict__ xlo, long n) {
  long t = (long)blockIdx.x * blockDim.x + threadIdx.x;
  if (t >= n) return;
  float v = X[t];
  unsigned short hi = f2bf(v);
  xhi[t] = hi;
  xlo[t] = f2bf(v - bf2f(hi));
}

// ---------- WMMA fragment loads ----------
// A 16x32 bf16: lane = M + 16*khalf;  K(j) = kb + khalf*8 + j (j<8), kb+16+khalf*8+j-8 (j>=8)
__device__ __forceinline__ v16bf load_fragA(const unsigned short* __restrict__ row_base,
                                            int kb, int khalf) {
  const unsigned short* p = row_base + kb + khalf * 8;
  v8us a = *reinterpret_cast<const v8us*>(p);        // 16B
  v8us b = *reinterpret_cast<const v8us*>(p + 16);   // 16B
  v16us u;
#pragma unroll
  for (int j = 0; j < 8; ++j) { u[j] = a[j]; u[j + 8] = b[j]; }
  return __builtin_bit_cast(v16bf, u);
}
// B 32x16 bf16: lane = N + 16*khalf;  K(j) = kb + khalf*8 + j (j<8), kb+16+khalf*8+j-8 (j>=8)
__device__ __forceinline__ v16bf load_fragB(const unsigned short* __restrict__ row_base,
                                            int kb, int khalf) {
  const unsigned short* p = row_base + kb + khalf * 8;
  v8us a = *reinterpret_cast<const v8us*>(p);        // 16B
  v8us b = *reinterpret_cast<const v8us*>(p + 16);   // 16B
  v16us u;
#pragma unroll
  for (int j = 0; j < 8; ++j) { u[j] = a[j]; u[j + 8] = b[j]; }
  return __builtin_bit_cast(v16bf, u);
}

// ---------- split-bf16 GEMM: H[N,128] = A[N,128] @ Wt^T (Wt is [n][k]) ----------
__global__ __launch_bounds__(128) void k_gemm(
    const unsigned short* __restrict__ Ahi, const unsigned short* __restrict__ Alo,
    const unsigned short* __restrict__ Bhi, const unsigned short* __restrict__ Blo,
    float* __restrict__ H, int nrows) {
  const int wave = threadIdx.x >> 5;
  const int lane = threadIdx.x & 31;
  const int rowBase = blockIdx.x * 64 + wave * 16;
  if (rowBase >= nrows) return;                 // wave-uniform exit
  const int r = lane & 15;
  const int khalf = lane >> 4;
  int arow = rowBase + r;
  if (arow > nrows - 1) arow = nrows - 1;       // clamp so EXEC stays all-1s

  v16bf ahi[4], alo[4];
#pragma unroll
  for (int kk = 0; kk < 4; ++kk) {
    ahi[kk] = load_fragA(Ahi + (long)arow * HDIM, kk * 32, khalf);
    alo[kk] = load_fragA(Alo + (long)arow * HDIM, kk * 32, khalf);
  }
#pragma unroll
  for (int nt = 0; nt < 8; ++nt) {
    const int ncol = nt * 16 + r;
    v8f acc = {};
#pragma unroll
    for (int kk = 0; kk < 4; ++kk) {
      v16bf bhi = load_fragB(Bhi + (long)ncol * HDIM, kk * 32, khalf);
      v16bf blo = load_fragB(Blo + (long)ncol * HDIM, kk * 32, khalf);
      acc = __builtin_amdgcn_wmma_f32_16x16x32_bf16(false, ahi[kk], false, bhi,
                                                    (short)0, acc, false, false);
      acc = __builtin_amdgcn_wmma_f32_16x16x32_bf16(false, ahi[kk], false, blo,
                                                    (short)0, acc, false, false);
      acc = __builtin_amdgcn_wmma_f32_16x16x32_bf16(false, alo[kk], false, bhi,
                                                    (short)0, acc, false, false);
    }
#pragma unroll
    for (int d = 0; d < 8; ++d) {
      int grow = rowBase + d + khalf * 8;
      if (grow < nrows) H[(long)grow * HDIM + ncol] = acc[d];
    }
  }
}

// ---------- edge scatter: agg[col] += dinv[row]*dinv[col] * H[row] ----------
__global__ void k_scatter(const int* __restrict__ row, const int* __restrict__ col,
                          const float* __restrict__ dinv, const float* __restrict__ H,
                          float* __restrict__ agg, int E) {
  long tid = (long)blockIdx.x * blockDim.x + threadIdx.x;
  long e = tid >> 5;
  if (e >= E) return;
  int c = ((int)tid & 31) * 4;
  int rs = row[e], cd = col[e];
  float w = dinv[rs] * dinv[cd];
  const float4 v = *reinterpret_cast<const float4*>(H + (long)rs * HDIM + c);
  float* dst = agg + (long)cd * HDIM + c;
  atomicAdd(dst + 0, v.x * w);
  atomicAdd(dst + 1, v.y * w);
  atomicAdd(dst + 2, v.z * w);
  atomicAdd(dst + 3, v.w * w);
}

// epilogue: self-loop term + bias (+ReLU), re-split to bf16 hi/lo for next layer
__global__ void k_post_split(const float* __restrict__ agg, const float* __restrict__ h,
                             const float* __restrict__ dinv, const float* __restrict__ bias,
                             unsigned short* __restrict__ ohi, unsigned short* __restrict__ olo,
                             long n) {
  long t = (long)blockIdx.x * blockDim.x + threadIdx.x;
  if (t >= n) return;
  int i = (int)(t >> 7), c = (int)(t & 127);
  float di = dinv[i];
  float v = agg[t] + di * di * h[t] + bias[c];
  v = fmaxf(v, 0.0f);
  unsigned short hi = f2bf(v);
  ohi[t] = hi;
  olo[t] = f2bf(v - bf2f(hi));
}
// last layer: no ReLU, keep f32 (written in place over h)
__global__ void k_post_final(const float* __restrict__ agg, float* __restrict__ h,
                             const float* __restrict__ dinv, const float* __restrict__ bias,
                             long n) {
  long t = (long)blockIdx.x * blockDim.x + threadIdx.x;
  if (t >= n) return;
  int i = (int)(t >> 7), c = (int)(t & 127);
  float di = dinv[i];
  h[t] = agg[t] + di * di * h[t] + bias[c];
}

// ---------- pooling + head ----------
__global__ void k_pool(const float* __restrict__ h3, const int* __restrict__ batch,
                       float* __restrict__ sums, float* __restrict__ cnts, long n) {
  long t = (long)blockIdx.x * blockDim.x + threadIdx.x;
  if (t >= n) return;
  int i = (int)(t >> 7), c = (int)(t & 127);
  int g = batch[i];
  atomicAdd(&sums[(long)g * HDIM + c], h3[t]);
  if (c == 0) atomicAdd(&cnts[g], 1.0f);
}
__global__ void k_pooled(float* __restrict__ sums, const float* __restrict__ cnts, int n) {
  int t = blockIdx.x * blockDim.x + threadIdx.x;
  if (t < n) sums[t] /= fmaxf(cnts[t >> 7], 1.0f);
}
__global__ void k_head1(const float* __restrict__ pooled, const float* __restrict__ lw1,
                        const float* __restrict__ lb1, float* __restrict__ z) {
  int t = blockIdx.x * blockDim.x + threadIdx.x;   // 64*128
  if (t >= GDIM * HDIM) return;
  int g = t >> 7, c = t & 127;
  float acc = lb1[c];
#pragma unroll 4
  for (int k = 0; k < HDIM; ++k) acc += pooled[g * HDIM + k] * lw1[k * HDIM + c];
  z[t] = fmaxf(acc, 0.0f);
}
__global__ void k_head2(const float* __restrict__ z, const float* __restrict__ lw2,
                        const float* __restrict__ lb2, float* __restrict__ out) {
  int t = blockIdx.x * blockDim.x + threadIdx.x;   // 64*10
  if (t >= GDIM * CDIM) return;
  int g = t / CDIM, c = t % CDIM;
  float acc = lb2[c];
#pragma unroll 4
  for (int k = 0; k < HDIM; ++k) acc += z[g * HDIM + k] * lw2[k * CDIM + c];
  out[t] = acc;
}

// ---------- driver ----------
extern "C" void kernel_launch(void* const* d_in, const int* in_sizes, int n_in,
                              void* d_out, int out_size, void* d_ws, size_t ws_size,
                              hipStream_t stream) {
  (void)n_in; (void)out_size; (void)ws_size;
  const float* x    = (const float*)d_in[0];
  const int*   edge = (const int*)d_in[1];
  const int*   batch= (const int*)d_in[2];
  const float* Ws[3] = {(const float*)d_in[3], (const float*)d_in[5], (const float*)d_in[7]};
  const float* bs[3] = {(const float*)d_in[4], (const float*)d_in[6], (const float*)d_in[8]};
  const float* lw1 = (const float*)d_in[9];
  const float* lb1 = (const float*)d_in[10];
  const float* lw2 = (const float*)d_in[11];
  const float* lb2 = (const float*)d_in[12];

  const int N = in_sizes[0] / HDIM;
  const int E = in_sizes[1] / 2;
  const int* erow = edge;
  const int* ecol = edge + E;
  const long nElem = (long)N * HDIM;

  // workspace carve-out (256B aligned): ~77 MB total
  size_t off = 0;
  char* base = (char*)d_ws;
  auto alloc = [&](size_t bytes) -> void* {
    off = (off + 255) & ~(size_t)255;
    void* p = base + off;
    off += bytes;
    return p;
  };
  float*          dinv = (float*)alloc((size_t)N * 4);
  unsigned short* xhi  = (unsigned short*)alloc((size_t)nElem * 2);
  unsigned short* xlo  = (unsigned short*)alloc((size_t)nElem * 2);
  float*          h    = (float*)alloc((size_t)nElem * 4);
  float*          agg  = (float*)alloc((size_t)nElem * 4);
  unsigned short* wthi[3]; unsigned short* wtlo[3];
  for (int l = 0; l < 3; ++l) {
    wthi[l] = (unsigned short*)alloc((size_t)HDIM * HDIM * 2);
    wtlo[l] = (unsigned short*)alloc((size_t)HDIM * HDIM * 2);
  }
  float* sums = (float*)alloc((size_t)GDIM * HDIM * 4);
  float* cnts = (float*)alloc((size_t)GDIM * 4);
  float* z    = (float*)alloc((size_t)GDIM * HDIM * 4);

  const int TB = 256;
  auto gr = [&](long n) { return (unsigned)((n + TB - 1) / TB); };

  // degrees -> dinv
  k_fill1<<<gr(N), TB, 0, stream>>>(dinv, N);
  k_deg  <<<gr(E), TB, 0, stream>>>(ecol, dinv, E);
  k_rsqrt<<<gr(N), TB, 0, stream>>>(dinv, N);

  // weights: transpose + split
  for (int l = 0; l < 3; ++l)
    k_splitW<<<gr(HDIM * HDIM), TB, 0, stream>>>(Ws[l], wthi[l], wtlo[l]);

  // features: split
  k_splitX<<<gr(nElem), TB, 0, stream>>>(x, xhi, xlo, nElem);

  // three GCN layers
  for (int l = 0; l < 3; ++l) {
    k_gemm<<<(unsigned)((N + 63) / 64), 128, 0, stream>>>(xhi, xlo, wthi[l], wtlo[l], h, N);
    k_zero<<<gr(nElem), TB, 0, stream>>>(agg, nElem);
    k_scatter<<<gr((long)E * 32), TB, 0, stream>>>(erow, ecol, dinv, h, agg, E);
    if (l < 2)
      k_post_split<<<gr(nElem), TB, 0, stream>>>(agg, h, dinv, bs[l], xhi, xlo, nElem);
    else
      k_post_final<<<gr(nElem), TB, 0, stream>>>(agg, h, dinv, bs[l], nElem);
  }

  // global mean pool
  k_zero<<<gr(GDIM * HDIM), TB, 0, stream>>>(sums, GDIM * HDIM);
  k_zero<<<gr(GDIM), TB, 0, stream>>>(cnts, GDIM);
  k_pool  <<<gr(nElem), TB, 0, stream>>>(h, batch, sums, cnts, nElem);
  k_pooled<<<gr(GDIM * HDIM), TB, 0, stream>>>(sums, cnts, GDIM * HDIM);

  // MLP head
  k_head1<<<gr(GDIM * HDIM), TB, 0, stream>>>(sums, lw1, lb1, z);
  k_head2<<<gr(GDIM * CDIM), TB, 0, stream>>>(z, lw2, lb2, (float*)d_out);
}
